// FlexibleResampler_89910845374837
// MI455X (gfx1250) — matrix operand order repaired and checked by reading, not compiled, not run on hardware
//
#include <hip/hip_runtime.h>
#include <math.h>

// ---------------------------------------------------------------------------
// FlexibleResampler forward for MI455X (gfx1250, wave32, WMMA).
// All GEMMs run on pre-converted f16 operands (A row-major [M,K], W transposed
// [N,K]) -> staging is pure b128 copies, inner loop is ds_load_b128 + wmma.
// Dominant cost: kv = LN(xp) @ Wkv (8 x [36928,1024]x[1024,2048] = 1.24 TF).
// ---------------------------------------------------------------------------

typedef __attribute__((ext_vector_type(16))) _Float16     v16h;
typedef __attribute__((ext_vector_type(8)))  float        v8f;
typedef __attribute__((ext_vector_type(4)))  unsigned int u32x4;

#define BM 128
#define BN 128
#define BK 64
#define SA 72   // LDS row stride in halfs (64 + 8 pad), 144B keeps 16B align
#define SB 72

enum { EPI_BIAS = 1, EPI_RESID = 2, EPI_GELU = 4, EPI_OUTF16 = 8 };

union FragU { v16h h; u32x4 u[2]; };

__device__ __forceinline__ float gelu_exact(float x) {
  return 0.5f * x * (1.0f + erff(x * 0.70710678118654752f));
}

// C = epilogue( A[M,K](f16) @ Wt[N,K](f16)^T ), f32 accum via v_wmma_f32_16x16x32_f16.
// Output row r scatters as C + (r/seg)*segstride + (r%seg)*ldc (kv concat).
// Requires K % 64 == 0, N % 128 == 0 (true for all GEMMs here).
__global__ __launch_bounds__(256)
void gemm_f16_wmma(const _Float16* __restrict__ A, int lda,
                   const _Float16* __restrict__ Wt,
                   float* __restrict__ C, int ldc,
                   int M, int K,
                   unsigned seg, unsigned long long segstride,
                   const float* __restrict__ bias, int flags)
{
  __shared__ __align__(16) _Float16 As[BM * SA];
  __shared__ __align__(16) _Float16 Bs[BN * SB];

  const int tid  = threadIdx.x;
  const int lane = tid & 31;
  const int wid  = tid >> 5;      // 8 waves: 2 (M) x 4 (N)
  const int wm   = wid >> 2;
  const int wn   = wid & 3;
  const int lmod = lane & 15;
  const int lhi  = lane >> 4;

  const int mbase = blockIdx.x * BM;
  const int nbase = blockIdx.y * BN;

  // Tile = 128 rows x 64 halfs = 1024 x 16B chunks; 4 chunks per thread/side.
  // chunk c: row = c>>3, koff = (c&7)*8 halfs.
  u32x4 ra[4], rb[4];

  const int nk = K / BK;

  // ---- load tile 0 into regs ----
  {
    for (int p = 0; p < 4; ++p) {
      int c = tid + p * 256;
      int row = c >> 3, ko = (c & 7) * 8;
      int gr = mbase + row;
      u32x4 v = {};
      if (gr < M) v = *(const u32x4*)(A + (size_t)gr * lda + ko);
      ra[p] = v;
      rb[p] = *(const u32x4*)(Wt + (size_t)(nbase + row) * K + ko);
    }
    for (int p = 0; p < 4; ++p) {
      int c = tid + p * 256;
      int row = c >> 3, ko = (c & 7) * 8;
      *(u32x4*)(As + row * SA + ko) = ra[p];
      *(u32x4*)(Bs + row * SB + ko) = rb[p];
    }
  }
  __syncthreads();

  v8f acc[4][2] = {};

  for (int kt = 0; kt < nk; ++kt) {
    // prefetch next K tile into regs while LDS tile is consumed
    if (kt + 1 < nk) {
      const int kb = (kt + 1) * BK;
      for (int p = 0; p < 4; ++p) {
        int c = tid + p * 256;
        int row = c >> 3, ko = (c & 7) * 8;
        int gr = mbase + row;
        u32x4 v = {};
        if (gr < M) v = *(const u32x4*)(A + (size_t)gr * lda + kb + ko);
        ra[p] = v;
        rb[p] = *(const u32x4*)(Wt + (size_t)(nbase + row) * K + kb + ko);
      }
      if (kt + 2 < nk) {               // -> global_prefetch_b8
        const int kb2 = (kt + 2) * BK;
        __builtin_prefetch(Wt + (size_t)(nbase + (tid >> 1)) * K + kb2, 0, 1);
        __builtin_prefetch(A + (size_t)(mbase + (tid >> 1)) * lda + kb2, 0, 1);
      }
    }

    // ---- 2 K-steps of 32 -> 16 WMMAs per barrier interval ----
    for (int s = 0; s < 2; ++s) {
      FragU fa[4], fb[2];
      for (int i = 0; i < 4; ++i) {
        // f16 A 16x32 layout: lanes 0-15 K={0..7,16..23}, lanes 16-31 K={8..15,24..31}
        const _Float16* pr = As + (wm * 64 + i * 16 + lmod) * SA + s * 32;
        fa[i].u[0] = *(const u32x4*)(pr + lhi * 8);
        fa[i].u[1] = *(const u32x4*)(pr + 16 + lhi * 8);
      }
      for (int j = 0; j < 2; ++j) {
        // B 32x16: lane = col; lanes 0-15 K={0..7,16..23}, lanes 16-31 K={8..15,24..31}
        const _Float16* pc = Bs + (wn * 32 + j * 16 + lmod) * SB + s * 32;
        fb[j].u[0] = *(const u32x4*)(pc + lhi * 8);
        fb[j].u[1] = *(const u32x4*)(pc + 16 + lhi * 8);
      }
      for (int i = 0; i < 4; ++i)
        for (int j = 0; j < 2; ++j)
          acc[i][j] = __builtin_amdgcn_wmma_f32_16x16x32_f16(
              false, fa[i].h, false, fb[j].h, (short)0, acc[i][j], false, false);
    }

    __syncthreads();
    if (kt + 1 < nk) {
      for (int p = 0; p < 4; ++p) {
        int c = tid + p * 256;
        int row = c >> 3, ko = (c & 7) * 8;
        *(u32x4*)(As + row * SA + ko) = ra[p];
        *(u32x4*)(Bs + row * SB + ko) = rb[p];
      }
      __syncthreads();
    }
  }

  // ---- epilogue: C layout VGPR r -> (M = base + lhi*8 + r, N = lane%16) ----
  for (int i = 0; i < 4; ++i) {
    int mrowbase = mbase + wm * 64 + i * 16 + lhi * 8;
    for (int j = 0; j < 2; ++j) {
      int n = nbase + wn * 32 + j * 16 + lmod;
      float bv = (flags & EPI_BIAS) ? bias[n] : 0.0f;
      for (int r = 0; r < 8; ++r) {
        int m = mrowbase + r;
        if (m < M) {
          unsigned um = (unsigned)m;
          size_t rowoff = (size_t)(um / seg) * (size_t)segstride +
                          (size_t)(um % seg) * (size_t)ldc;
          float v = acc[i][j][r] + bv;
          if (flags & EPI_GELU) v = gelu_exact(v);
          if (flags & EPI_OUTF16) {
            ((_Float16*)C)[rowoff + n] = (_Float16)v;
          } else {
            float* cp = C + rowoff + n;
            if (flags & EPI_RESID) v += *cp;
            *cp = v;
          }
        }
      }
    }
  }
}

// ---------------------------------------------------------------------------
// LayerNorm: one row per 256-thread block; output type templated (f16 feeds
// the next GEMM, f32 for the final output).
// ---------------------------------------------------------------------------
template <typename OT>
__global__ __launch_bounds__(256)
void layernorm_kernel(const float* __restrict__ x, const float* __restrict__ g,
                      const float* __restrict__ b, OT* __restrict__ y, int D)
{
  __shared__ float red[16];
  const int row = blockIdx.x;
  const float* xr = x + (size_t)row * D;
  float s = 0.f, s2 = 0.f;
  for (int i = threadIdx.x; i < D; i += 256) { float v = xr[i]; s += v; s2 += v * v; }
  for (int o = 16; o > 0; o >>= 1) { s += __shfl_xor(s, o, 32); s2 += __shfl_xor(s2, o, 32); }
  const int w = threadIdx.x >> 5;
  if ((threadIdx.x & 31) == 0) { red[w] = s; red[w + 8] = s2; }
  __syncthreads();
  float ts = 0.f, ts2 = 0.f;
  for (int i = 0; i < 8; ++i) { ts += red[i]; ts2 += red[i + 8]; }
  const float mu  = ts / (float)D;
  const float var = ts2 / (float)D - mu * mu;
  const float rinv = rsqrtf(var + 1e-5f);
  OT* yr = y + (size_t)row * D;
  for (int i = threadIdx.x; i < D; i += 256)
    yr[i] = (OT)((xr[i] - mu) * rinv * g[i] + b[i]);
}

// f32 -> f16 flat convert
__global__ void cvt_f32_to_f16(const float* __restrict__ in, _Float16* __restrict__ o,
                               long long n)
{
  long long i = (long long)blockIdx.x * 256 + threadIdx.x;
  if (i < n) o[i] = (_Float16)in[i];
}

// Wt[n*K + k] = (f16) W[k*N + n]   (32x32 LDS tile transpose; K,N % 32 == 0)
__global__ __launch_bounds__(256)
void transpose_cvt_f16(const float* __restrict__ W, _Float16* __restrict__ Wt,
                       int K, int N)
{
  __shared__ float t[32][33];
  const int kb = blockIdx.x * 32, nb = blockIdx.y * 32;
  const int tx = threadIdx.x & 31, ty = threadIdx.x >> 5;
  for (int i = ty; i < 32; i += 8)
    t[i][tx] = W[(size_t)(kb + i) * N + nb + tx];
  __syncthreads();
  for (int i = ty; i < 32; i += 8)
    Wt[(size_t)(nb + i) * K + kb + tx] = (_Float16)t[tx][i];
}

// latents0 = broadcast(queries)
__global__ void bcast_latents(const float* __restrict__ q, float* __restrict__ lat,
                              int per, int total)
{
  int i = blockIdx.x * 256 + threadIdx.x;
  if (i < total) lat[i] = q[i % per];
}

// ---------------------------------------------------------------------------
// Attention: one wave per (batch, head, query). L=593 keys, dim_head=64.
// logits = (q.k) * 64^-0.5, softmax fp32, o = P@V written as f16 (feeds GEMM).
// ---------------------------------------------------------------------------
__global__ __launch_bounds__(32)
void attn_kernel(const float* __restrict__ qbuf, const float* __restrict__ kvbuf,
                 _Float16* __restrict__ obuf)
{
  const int L = 593;
  const int id = blockIdx.x;
  const int b  = id >> 8;
  const int h  = (id >> 4) & 15;
  const int qi = id & 15;
  const int lane = threadIdx.x;

  __shared__ float qs[64];
  __shared__ float ls[608];

  const float* qp = qbuf + ((size_t)(b * 16 + qi)) * 1024 + h * 64;
  qs[lane]      = qp[lane];
  qs[lane + 32] = qp[lane + 32];
  __syncthreads();

  const float* kbase = kvbuf + (size_t)b * 593 * 2048 + h * 64;
  float m = -1e30f;
  for (int k = lane; k < L; k += 32) {
    const float* kr = kbase + (size_t)k * 2048;
    float d = 0.f;
    for (int t = 0; t < 64; ++t) d += qs[t] * kr[t];
    d *= 0.125f;
    ls[k] = d;
    m = fmaxf(m, d);
  }
  for (int o = 16; o > 0; o >>= 1) m = fmaxf(m, __shfl_xor(m, o, 32));
  float sum = 0.f;
  for (int k = lane; k < L; k += 32) { float e = __expf(ls[k] - m); ls[k] = e; sum += e; }
  for (int o = 16; o > 0; o >>= 1) sum += __shfl_xor(sum, o, 32);
  const float inv = 1.0f / sum;
  __syncthreads();

  const float* vbase = kvbuf + (size_t)b * 593 * 2048 + 1024 + h * 64;
  float a0 = 0.f, a1 = 0.f;
  for (int k = 0; k < L; ++k) {
    float p = ls[k];
    const float* vr = vbase + (size_t)k * 2048;
    a0 += p * vr[lane];
    a1 += p * vr[lane + 32];
  }
  _Float16* op = obuf + ((size_t)(b * 16 + qi)) * 1024 + h * 64;
  op[lane]      = (_Float16)(a0 * inv);
  op[lane + 32] = (_Float16)(a1 * inv);
}

// ---------------------------------------------------------------------------
extern "C" void kernel_launch(void* const* d_in, const int* in_sizes, int n_in,
                              void* d_out, int out_size, void* d_ws, size_t ws_size,
                              hipStream_t stream)
{
  (void)in_sizes; (void)n_in; (void)out_size; (void)ws_size;

  const float* x    = (const float*)d_in[0];   // (64,577,768)
  const float* qrs  = (const float*)d_in[1];   // (1,16,1024)
  const float* piw  = (const float*)d_in[2];   // (768,1024)
  const float* pib  = (const float*)d_in[3];
  const float* ln1g = (const float*)d_in[4];
  const float* ln1b = (const float*)d_in[5];
  const float* ln2g = (const float*)d_in[6];
  const float* ln2b = (const float*)d_in[7];
  const float* wq   = (const float*)d_in[8];   // (8,1024,1024)
  const float* wkv  = (const float*)d_in[9];   // (8,1024,2048)
  const float* wo   = (const float*)d_in[10];  // (8,1024,1024)
  const float* ffg  = (const float*)d_in[11];
  const float* ffb  = (const float*)d_in[12];
  const float* w1   = (const float*)d_in[13];  // (8,1024,4096)
  const float* w2   = (const float*)d_in[14];  // (8,4096,1024)
  const float* poww = (const float*)d_in[15];  // (1024,1024)
  const float* pob  = (const float*)d_in[16];
  const float* ng   = (const float*)d_in[17];
  const float* nb   = (const float*)d_in[18];
  float* out = (float*)d_out;

  const int Mx = 64 * 577;   // 36928
  const int Ml = 64 * 16;    // 1024

  // ---- workspace carve-up (byte allocator, 256B aligned) ----
  char* base = (char*)d_ws;
  size_t off = 0;
  auto takeB = [&](size_t bytes) -> char* {
    char* p = base + off;
    off += (bytes + 255) & ~(size_t)255;
    return p;
  };
  // fp32 buffers
  float* xp   = (float*)takeB((size_t)Mx * 1024 * 4);
  float* kv   = (float*)takeB((size_t)64 * 593 * 2048 * 4);
  float* lat  = (float*)takeB((size_t)Ml * 1024 * 4);
  float* qb   = (float*)takeB((size_t)Ml * 1024 * 4);
  float* otmp = (float*)takeB((size_t)Ml * 1024 * 4);
  // f16 activation buffers
  _Float16* xf   = (_Float16*)takeB((size_t)Mx * 768 * 2);
  _Float16* xnf  = (_Float16*)takeB((size_t)Mx * 1024 * 2);
  _Float16* ltf  = (_Float16*)takeB((size_t)Ml * 1024 * 2);
  _Float16* obf  = (_Float16*)takeB((size_t)Ml * 1024 * 2);
  _Float16* ffnf = (_Float16*)takeB((size_t)Ml * 1024 * 2);
  _Float16* hbf  = (_Float16*)takeB((size_t)Ml * 4096 * 2);
  _Float16* latf = (_Float16*)takeB((size_t)Ml * 1024 * 2);
  // f16 transposed weights [N,K]
  _Float16* piwT = (_Float16*)takeB((size_t)768 * 1024 * 2);
  _Float16* wqT  = (_Float16*)takeB((size_t)8 * 1024 * 1024 * 2);
  _Float16* wkvT = (_Float16*)takeB((size_t)8 * 1024 * 2048 * 2);
  _Float16* woT  = (_Float16*)takeB((size_t)8 * 1024 * 1024 * 2);
  _Float16* w1T  = (_Float16*)takeB((size_t)8 * 1024 * 4096 * 2);
  _Float16* w2T  = (_Float16*)takeB((size_t)8 * 4096 * 1024 * 2);
  _Float16* powT = (_Float16*)takeB((size_t)1024 * 1024 * 2);

  auto xpose = [&](const float* W, _Float16* Wt, int K, int N) {
    transpose_cvt_f16<<<dim3(K / 32, N / 32), 256, 0, stream>>>(W, Wt, K, N);
  };
  auto gemm = [&](const _Float16* A, int lda, const _Float16* Wt, int N,
                  float* C, int ldc, int M, int K,
                  unsigned seg, unsigned long long segstride,
                  const float* bias, int flags) {
    dim3 g((unsigned)((M + BM - 1) / BM), (unsigned)(N / BN));
    gemm_f16_wmma<<<g, dim3(256), 0, stream>>>(A, lda, Wt, C, ldc, M, K,
                                               seg, segstride, bias, flags);
  };

  // ---- one-time (per launch) operand preparation ----
  {
    long long nx = (long long)Mx * 768;
    cvt_f32_to_f16<<<(unsigned)((nx + 255) / 256), 256, 0, stream>>>(x, xf, nx);
  }
  xpose(piw, piwT, 768, 1024);
  for (int l = 0; l < 8; ++l) {
    xpose(wq  + (size_t)l * 1024 * 1024, wqT  + (size_t)l * 1024 * 1024, 1024, 1024);
    xpose(wkv + (size_t)l * 1024 * 2048, wkvT + (size_t)l * 1024 * 2048, 1024, 2048);
    xpose(wo  + (size_t)l * 1024 * 1024, woT  + (size_t)l * 1024 * 1024, 1024, 1024);
    xpose(w1  + (size_t)l * 1024 * 4096, w1T  + (size_t)l * 1024 * 4096, 1024, 4096);
    xpose(w2  + (size_t)l * 4096 * 1024, w2T  + (size_t)l * 4096 * 1024, 4096, 1024);
  }
  xpose(poww, powT, 1024, 1024);

  // ---- forward ----
  gemm(xf, 768, piwT, 1024, xp, 1024, Mx, 768, (unsigned)Mx, 0ULL, pib, EPI_BIAS);
  {
    int total = Ml * 1024;
    bcast_latents<<<(total + 255) / 256, 256, 0, stream>>>(qrs, lat, 16 * 1024, total);
  }

  const unsigned long long kvstride = 593ULL * 2048ULL;
  for (int l = 0; l < 8; ++l) {
    layernorm_kernel<_Float16><<<Mx, 256, 0, stream>>>(xp, ln1g + l * 1024, ln1b + l * 1024, xnf, 1024);
    layernorm_kernel<_Float16><<<Ml, 256, 0, stream>>>(lat, ln2g + l * 1024, ln2b + l * 1024, ltf, 1024);
    // q = lt @ Wq  (fp32 out for attention)
    gemm(ltf, 1024, wqT + (size_t)l * 1024 * 1024, 1024, qb, 1024, Ml, 1024,
         (unsigned)Ml, 0ULL, nullptr, 0);
    // kv rows 0..576 per batch: xn @ Wkv  (row remap r -> (r/577, r%577))
    gemm(xnf, 1024, wkvT + (size_t)l * 1024 * 2048, 2048, kv, 2048, Mx, 1024,
         577u, kvstride, nullptr, 0);
    // kv rows 577..592 per batch: lt @ Wkv
    gemm(ltf, 1024, wkvT + (size_t)l * 1024 * 2048, 2048, kv + (size_t)577 * 2048, 2048,
         Ml, 1024, 16u, kvstride, nullptr, 0);
    attn_kernel<<<64 * 16 * 16, 32, 0, stream>>>(qb, kv, obf);
    // latents += o @ Wo
    gemm(obf, 1024, woT + (size_t)l * 1024 * 1024, 1024, lat, 1024, Ml, 1024,
         (unsigned)Ml, 0ULL, nullptr, EPI_RESID);
    // FF
    layernorm_kernel<_Float16><<<Ml, 256, 0, stream>>>(lat, ffg + l * 1024, ffb + l * 1024, ffnf, 1024);
    gemm(ffnf, 1024, w1T + (size_t)l * 1024 * 4096, 4096, (float*)hbf, 4096, Ml, 1024,
         (unsigned)Ml, 0ULL, nullptr, EPI_GELU | EPI_OUTF16);
    gemm(hbf, 4096, w2T + (size_t)l * 4096 * 1024, 1024, lat, 1024, Ml, 4096,
         (unsigned)Ml, 0ULL, nullptr, EPI_RESID);
  }

  // out = LN(latents @ proj_out_w + proj_out_b)
  {
    long long nl = (long long)Ml * 1024;
    cvt_f32_to_f16<<<(unsigned)((nl + 255) / 256), 256, 0, stream>>>(lat, latf, nl);
  }
  gemm(latf, 1024, powT, 1024, otmp, 1024, Ml, 1024, (unsigned)Ml, 0ULL, pob, EPI_BIAS);
  layernorm_kernel<float><<<Ml, 256, 0, stream>>>(otmp, ng, nb, out, 1024);
}
